// attention_42125039239258
// MI455X (gfx1250) — matrix operand order repaired and checked by reading, not compiled, not run on hardware
//
#include <hip/hip_runtime.h>
#include <hip/hip_bf16.h>
#include <math.h>

typedef __bf16 bf16_t;
typedef __attribute__((ext_vector_type(16))) __bf16 v16bf;
typedef __attribute__((ext_vector_type(8)))  __bf16 v8bf;
typedef __attribute__((ext_vector_type(8)))  float  v8f;
typedef __attribute__((ext_vector_type(4)))  float  v4f;

#define E2V 1024
#define DV  1024
#define NBATCH 16
#define SLEN 2048
#define APAD 8
#define ASTRIDE (E2V + APAD)   // 1032 bf16 per LDS row (2064 B -> bank-spread)

// ---------------------------------------------------------------------------
// K1: W1k (f32, [k][n] row-major) -> W1kT (bf16, [n][k] row-major), tiled
// ---------------------------------------------------------------------------
__global__ void k_w1k_transpose(const float* __restrict__ W1, bf16_t* __restrict__ W1kT) {
    __shared__ float tile[32][33];
    int tx = threadIdx.x & 31, ty = threadIdx.x >> 5;       // 32 x 8
    int nb = blockIdx.x, kb = blockIdx.y;
    int n = nb * 32 + tx;
    #pragma unroll
    for (int i = 0; i < 32; i += 8) {
        int k = kb * 32 + ty + i;
        tile[ty + i][tx] = W1[(size_t)k * DV + n];          // coalesced in n
    }
    __syncthreads();
    int k2 = kb * 32 + tx;
    #pragma unroll
    for (int i = 0; i < 32; i += 8) {
        int n2 = nb * 32 + ty + i;
        W1kT[(size_t)n2 * E2V + k2] = (bf16_t)tile[tx][ty + i];  // coalesced in k
    }
}

// ---------------------------------------------------------------------------
// K2: dprojb[n][d] = sum_e dec[n][e] * W1d[e][d] + b1[d]
// ---------------------------------------------------------------------------
__global__ void k_decproj(const float* __restrict__ dec, const float* __restrict__ W1,
                          const float* __restrict__ b1, float* __restrict__ dprojb) {
    int n = blockIdx.x, tid = threadIdx.x;
    __shared__ float sdec[DV];
    for (int i = tid; i < DV; i += 256) sdec[i] = dec[n * DV + i];
    __syncthreads();
    const float* W1d = W1 + (size_t)E2V * DV;
    float acc[4] = {0.f, 0.f, 0.f, 0.f};
    for (int e = 0; e < DV; ++e) {
        float dv = sdec[e];
        const float* rowp = W1d + (size_t)e * DV;
        #pragma unroll
        for (int j = 0; j < 4; ++j) acc[j] = fmaf(dv, rowp[tid + j * 256], acc[j]);
    }
    #pragma unroll
    for (int j = 0; j < 4; ++j) {
        int d = tid + j * 256;
        dprojb[n * DV + d] = acc[j] + b1[d];
    }
}

// ---------------------------------------------------------------------------
// K3: main fused kernel. One block per 16-row strip of K (2048 blocks).
//   - stage K strip -> LDS (bf16) + fused f32 copy to out3
//   - 8 waves x 8 N-tiles: v_wmma_f32_16x16x32_bf16, K-loop of 32 steps
//   - epilogue: energy[row] = sum_n tanh(acc + dprojb)*W2  (+ b2)
// ---------------------------------------------------------------------------
__global__ void __launch_bounds__(256) k_main(
    const float* __restrict__ Kin, const bf16_t* __restrict__ W1kT,
    const float* __restrict__ dprojb, const float* __restrict__ W2,
    const float* __restrict__ b2, float* __restrict__ outK,
    float* __restrict__ energy) {
    __shared__ bf16_t ldsA[16 * ASTRIDE];
    __shared__ float  lds_dp[DV];
    __shared__ float  lds_w2[DV];
    __shared__ float  lds_e[16];

    int tid    = threadIdx.x;
    int row0   = blockIdx.x * 16;        // global row (n*S + s) of strip start
    int nbatch = row0 / SLEN;            // strips never straddle batches (2048%16==0)

    // ---- stage A strip (16 x 1024 f32 -> bf16 LDS) + fused copy to out3 ----
    #pragma unroll
    for (int i = 0; i < 16; ++i) {
        int idx = tid + i * 256;         // 0..4095 float4 groups
        int off = idx * 4;
        int row = off >> 10, kc = off & 1023;
        v4f f = *(const v4f*)(Kin + (size_t)(row0 + row) * E2V + kc);
        *(v4f*)(outK + (size_t)row0 * E2V + off) = f;       // fused key copy
        union { bf16_t b[4]; unsigned long long u; } p;
        p.b[0] = (bf16_t)f.x; p.b[1] = (bf16_t)f.y;
        p.b[2] = (bf16_t)f.z; p.b[3] = (bf16_t)f.w;
        *(unsigned long long*)(ldsA + row * ASTRIDE + kc) = p.u;
    }
    for (int i = tid; i < DV; i += 256) {
        lds_dp[i] = dprojb[nbatch * DV + i];
        lds_w2[i] = W2[i];
    }
    if (tid < 16) lds_e[tid] = 0.f;
    __syncthreads();

    // ---- WMMA main loop ----
    int wave = tid >> 5, lane = tid & 31;
    int half = lane >> 4, l16 = lane & 15;
    int n0base = wave * 128;             // each wave covers 128 output columns

    v8f c[8] = {};
    const bf16_t* ldsRow = ldsA + l16 * ASTRIDE;   // A: lane -> row M=l16

    for (int ks = 0; ks < 32; ++ks) {
        int kb = ks * 32;
        // A frag: half-wave h holds K = kb+h*16..+15 (matches B)
        union { v16bf v; struct { v8bf lo, hi; } h; } a;
        a.h.lo = *(const v8bf*)(ldsRow + kb + half * 16);
        a.h.hi = *(const v8bf*)(ldsRow + kb + half * 16 + 8);
        #pragma unroll
        for (int t = 0; t < 8; ++t) {
            int col = n0base + t * 16 + l16;       // B: lane -> column N
            // B frag: 16 consecutive K of one column, half-wave h -> K = kb+h*16..+15
            v16bf b = *(const v16bf*)(W1kT + (size_t)col * E2V + kb + half * 16);
            c[t] = __builtin_amdgcn_wmma_f32_16x16x32_bf16(
                       false, a.v, false, b, (short)0, c[t], false, false);
        }
    }

    // ---- epilogue: tanh + dot with W2, reduce to per-row energy ----
    float ep[8];
    #pragma unroll
    for (int v = 0; v < 8; ++v) ep[v] = 0.f;
    #pragma unroll
    for (int t = 0; t < 8; ++t) {
        int col = n0base + t * 16 + l16;
        float dp = lds_dp[col];
        float w2 = lds_w2[col];
        #pragma unroll
        for (int v = 0; v < 8; ++v) {            // C layout: row M = v + 8*half
            float h = tanhf(c[t][v] + dp);
            ep[v] = fmaf(h, w2, ep[v]);
        }
    }
    #pragma unroll
    for (int v = 0; v < 8; ++v) {
        ep[v] += __shfl_xor(ep[v], 8, 16);
        ep[v] += __shfl_xor(ep[v], 4, 16);
        ep[v] += __shfl_xor(ep[v], 2, 16);
        ep[v] += __shfl_xor(ep[v], 1, 16);
    }
    if (l16 == 0) {
        #pragma unroll
        for (int v = 0; v < 8; ++v) atomicAdd(&lds_e[v + 8 * half], ep[v]);
    }
    __syncthreads();
    if (tid < 16) {
        int s = (row0 % SLEN) + tid;
        energy[nbatch * SLEN + s] = lds_e[tid] + b2[0];
    }
}

// ---------------------------------------------------------------------------
// K4: softmax over S per batch, apply masks -> scores (out1)
// ---------------------------------------------------------------------------
__global__ void k_softmax(const float* __restrict__ energy, const float* __restrict__ amask,
                          const float* __restrict__ imask, float* __restrict__ scores) {
    int n = blockIdx.x, tid = threadIdx.x;
    __shared__ float red[256];
    float le[8];
    float lmax = -INFINITY;
    #pragma unroll
    for (int i = 0; i < 8; ++i) {
        int s = tid + i * 256;
        le[i] = energy[n * SLEN + s] + amask[n * SLEN + s];
        lmax = fmaxf(lmax, le[i]);
    }
    red[tid] = lmax; __syncthreads();
    for (int o = 128; o > 0; o >>= 1) {
        if (tid < o) red[tid] = fmaxf(red[tid], red[tid + o]);
        __syncthreads();
    }
    float m = red[0]; __syncthreads();
    float lsum = 0.f;
    #pragma unroll
    for (int i = 0; i < 8; ++i) { le[i] = __expf(le[i] - m); lsum += le[i]; }
    red[tid] = lsum; __syncthreads();
    for (int o = 128; o > 0; o >>= 1) {
        if (tid < o) red[tid] += red[tid + o];
        __syncthreads();
    }
    float inv = 1.0f / red[0];
    #pragma unroll
    for (int i = 0; i < 8; ++i) {
        int s = tid + i * 256;
        scores[n * SLEN + s] = le[i] * inv * imask[n * SLEN + s];
    }
}

// ---------------------------------------------------------------------------
// K5/K6: context = sum_s scores * V, deterministic two-phase reduction
// ---------------------------------------------------------------------------
__global__ void k_ctx_partial(const float* __restrict__ scores, const float* __restrict__ V,
                              float* __restrict__ part) {
    int n = blockIdx.x >> 4, chunk = blockIdx.x & 15;
    int d4 = threadIdx.x * 4;
    v4f acc = {};
    int s0 = chunk * 128;
    for (int s = s0; s < s0 + 128; ++s) {
        float w = scores[n * SLEN + s];
        v4f v = *(const v4f*)(V + ((size_t)n * SLEN + s) * DV + d4);
        acc += w * v;
    }
    *(v4f*)(part + (size_t)blockIdx.x * DV + d4) = acc;
}

__global__ void k_ctx_reduce(const float* __restrict__ part, float* __restrict__ ctx) {
    int idx = blockIdx.x * 256 + threadIdx.x;   // 0..16383
    int n = idx >> 10, d = idx & 1023;
    float s = 0.f;
    #pragma unroll
    for (int c = 0; c < 16; ++c) s += part[(size_t)(n * 16 + c) * DV + d];
    ctx[idx] = s;
}

// ---------------------------------------------------------------------------
extern "C" void kernel_launch(void* const* d_in, const int* in_sizes, int n_in,
                              void* d_out, int out_size, void* d_ws, size_t ws_size,
                              hipStream_t stream) {
    (void)in_sizes; (void)n_in; (void)out_size; (void)ws_size;
    const float* Kin   = (const float*)d_in[0];
    const float* V     = (const float*)d_in[1];
    const float* dec   = (const float*)d_in[2];
    const float* amask = (const float*)d_in[3];
    const float* imask = (const float*)d_in[4];
    const float* W1    = (const float*)d_in[5];
    const float* b1    = (const float*)d_in[6];
    const float* W2    = (const float*)d_in[7];
    const float* b2    = (const float*)d_in[8];

    float* out        = (float*)d_out;
    float* out_scores = out;                               // 16*2048
    float* out_ctx    = out + NBATCH * SLEN;               // 16*1024
    float* out_K      = out + NBATCH * SLEN + NBATCH * DV; // 16*2048*1024 copy

    char*   ws     = (char*)d_ws;
    bf16_t* W1kT   = (bf16_t*)ws;                                   // 2 MB
    float*  dprojb = (float*)(ws + (size_t)2 * 1024 * 1024);        // 64 KB
    float*  energy = dprojb + NBATCH * DV;                          // 128 KB
    float*  part   = energy + NBATCH * SLEN;                        // 1 MB

    k_w1k_transpose<<<dim3(32, 32), 256, 0, stream>>>(W1, W1kT);
    k_decproj<<<NBATCH, 256, 0, stream>>>(dec, W1, b1, dprojb);
    k_main<<<(NBATCH * SLEN) / 16, 256, 0, stream>>>(Kin, W1kT, dprojb, W2, b2, out_K, energy);
    k_softmax<<<NBATCH, 256, 0, stream>>>(energy, amask, imask, out_scores);
    k_ctx_partial<<<NBATCH * 16, 256, 0, stream>>>(out_scores, V, part);
    k_ctx_reduce<<<(NBATCH * DV) / 256, 256, 0, stream>>>(part, out_ctx);
}
